// TransformerEncoder_3444563771645
// MI455X (gfx1250) — matrix operand order repaired and checked by reading, not compiled, not run on hardware
//
#include <hip/hip_runtime.h>
#include <hip/hip_bf16.h>
#include <stdint.h>

#define D_MODEL 1024
#define N_HEADS 16
#define D_KH    64
#define D_FF    4096
#define SEQ     2048
#define BATCH   2
#define SROWS   (BATCH * SEQ)   // 4096 token rows

typedef unsigned short u16t;
typedef __bf16    v16bf __attribute__((ext_vector_type(16)));
typedef float     v8f   __attribute__((ext_vector_type(8)));
typedef uint32_t  v4u   __attribute__((ext_vector_type(4)));
typedef uint32_t  v8u   __attribute__((ext_vector_type(8)));

union AB32 { v16bf v; uint4 q[2]; u16t h[16]; };

__device__ __forceinline__ u16t f2bf(float x) {
    uint32_t b = __float_as_uint(x);
    return (u16t)((b + 0x7FFFu + ((b >> 16) & 1u)) >> 16);
}

// Load one 16-element bf16 fragment from two contiguous 16B runs.
__device__ __forceinline__ v16bf ld_frag(const u16t* p0, const u16t* p1) {
    AB32 f;
    f.q[0] = *(const uint4*)p0;
    f.q[1] = *(const uint4*)p1;
    return f.v;
}

// ---------------------------------------------------------------------------
// Tensor Data Mover: 2D tile (bf16) global -> LDS.
// D# per CDNA5 ISA:
//   group0: [1:0]=count=1, [63:32]=lds_addr, [120:64]=global_addr, [127:126]=2
//   group1: data_size=1(2B), tensor_dim0/1, tile_dim0/1, tensor_dim0_stride
// Issued wave-uniform; tracked by TENSORcnt (s_wait_tensorcnt).
// ---------------------------------------------------------------------------
__device__ __forceinline__ void tdm_load_2d(uint32_t lds_addr, const u16t* gaddr,
                                            uint32_t tensor_d0, uint32_t tensor_d1,
                                            uint32_t tile_d0, uint32_t tile_d1,
                                            uint32_t stride_d0)
{
    const uint64_t ga = (uint64_t)(uintptr_t)gaddr;
    v4u g0;
    g0[0] = 1u;                                   // count=1, user descriptor
    g0[1] = lds_addr;                             // LDS byte address
    g0[2] = (uint32_t)ga;                         // global_addr[31:0]
    g0[3] = (uint32_t)(ga >> 32) | (2u << 30);    // global_addr[56:32] | type=2
    v8u g1;
    g1[0] = 0x10000u;                             // wg_mask=0, data_size=2B
    g1[1] = (tensor_d0 & 0xFFFFu) << 16;          // tensor_dim0[15:0]
    g1[2] = (tensor_d0 >> 16) | ((tensor_d1 & 0xFFFFu) << 16);
    g1[3] = (tensor_d1 >> 16) | (tile_d0 << 16);  // tensor_dim1[31:16] | tile_dim0
    g1[4] = tile_d1;                              // tile_dim1 (tile_dim2 = 0)
    g1[5] = stride_d0;                            // tensor_dim0_stride[31:0]
    g1[6] = 0u;
    g1[7] = 0u;
    asm volatile("tensor_load_to_lds %0, %1" :: "s"(g0), "s"(g1) : "memory");
}

// ---------------------------------------------------------------------------
// fp32 -> bf16 converter (vectorized x4)
// ---------------------------------------------------------------------------
__global__ void k_cvt4(const float* __restrict__ in, u16t* __restrict__ out, int n4) {
    int i = blockIdx.x * blockDim.x + threadIdx.x;
    if (i < n4) {
        float4 v = *(const float4*)(in + (size_t)i * 4);
        uint2 pk;
        pk.x = (uint32_t)f2bf(v.x) | ((uint32_t)f2bf(v.y) << 16);
        pk.y = (uint32_t)f2bf(v.z) | ((uint32_t)f2bf(v.w) << 16);
        *(uint2*)(out + (size_t)i * 4) = pk;
    }
}

// ---------------------------------------------------------------------------
// fp32 (KxN) -> bf16 transposed (NxK), 32x32 LDS tiles. One-time per weight.
// ---------------------------------------------------------------------------
__global__ __launch_bounds__(256)
void k_cvt_t(const float* __restrict__ in, u16t* __restrict__ out, int K, int N) {
    __shared__ u16t tile[32][33];
    const int kb = blockIdx.y * 32, nb = blockIdx.x * 32;
    const int tx = threadIdx.x & 31, ty = threadIdx.x >> 5;   // 8 rows of 32
    #pragma unroll
    for (int i = 0; i < 4; ++i)
        tile[ty + i * 8][tx] = f2bf(in[(size_t)(kb + ty + i * 8) * N + nb + tx]);
    __syncthreads();
    #pragma unroll
    for (int i = 0; i < 4; ++i)
        out[(size_t)(nb + ty + i * 8) * K + kb + tx] = tile[tx][ty + i * 8];
}

// ---------------------------------------------------------------------------
// bf16 WMMA GEMM: C(MxN, f32 accum) = A(MxK) @ Bt(NxK)^T + bias
// Block tile 128x128x32, 256 threads = 8 waves, wave tile 64x32.
// Both tiles staged by TDM (tensor_load_to_lds), double-buffered LDS;
// wave 0 issues descriptors for tile t+1 while all waves compute tile t.
// EPI: 0 = f32 store; 1 = bf16 store scaled; 2 = exact GELU -> bf16;
//      3 = bf16 store head-transposed for attention V ([b,h][dk][l]).
// ---------------------------------------------------------------------------
template <int EPI>
__global__ __launch_bounds__(256)
void k_gemm_bf16(const u16t* __restrict__ A, const u16t* __restrict__ Bt,
                 const float* __restrict__ bias,
                 float* __restrict__ Cf, u16t* __restrict__ Cb,
                 int M, int N, int K, float scale)
{
    __shared__ u16t sA[2][128 * 32];   // [m][k]
    __shared__ u16t sB[2][128 * 32];   // [n][k]

    const int t    = threadIdx.x;
    const int lane = t & 31;
    const int w    = t >> 5;
    const int wr   = w >> 2;            // 0..1
    const int wc   = w & 3;             // 0..3
    const int mBase = blockIdx.y * 128;
    const int nBase = blockIdx.x * 128;
    const int mW = wr * 64, nW = wc * 32;
    const int l15 = lane & 15;
    const int hiA = (lane < 16) ? 0 : 8;
    const int hiB = (lane < 16) ? 0 : 8;
    const int rAdd = (lane < 16) ? 0 : 8;

    v8f acc[4][2];
    #pragma unroll
    for (int i = 0; i < 4; ++i)
        #pragma unroll
        for (int j = 0; j < 2; ++j)
            acc[i][j] = (v8f){0.f, 0.f, 0.f, 0.f, 0.f, 0.f, 0.f, 0.f};

    const u16t* Ag = A  + (size_t)mBase * K;
    const u16t* Bg = Bt + (size_t)nBase * K;
    const int nt = K >> 5;

    if (w == 0) {
        tdm_load_2d((uint32_t)(uintptr_t)&sA[0][0], Ag, K, M, 32, 128, K);
        tdm_load_2d((uint32_t)(uintptr_t)&sB[0][0], Bg, K, N, 32, 128, K);
    }

    for (int kt = 0; kt < nt; ++kt) {
        const int cur = kt & 1;
        if (w == 0) {
            if (kt + 1 < nt) {
                const int k0n = (kt + 1) << 5;
                tdm_load_2d((uint32_t)(uintptr_t)&sA[cur ^ 1][0], Ag + k0n, K, M, 32, 128, K);
                tdm_load_2d((uint32_t)(uintptr_t)&sB[cur ^ 1][0], Bg + k0n, K, N, 32, 128, K);
                __builtin_amdgcn_s_wait_tensorcnt(2);   // tile kt resident
            } else {
                __builtin_amdgcn_s_wait_tensorcnt(0);
            }
        }
        __syncthreads();

        v16bf af[4], bfr[2];
        #pragma unroll
        for (int mf = 0; mf < 4; ++mf) {
            const u16t* p = sA[cur] + (mW + mf * 16 + l15) * 32;
            af[mf] = ld_frag(p + hiA, p + 16 + hiA);
        }
        #pragma unroll
        for (int nf = 0; nf < 2; ++nf) {
            const u16t* p = sB[cur] + (nW + nf * 16 + l15) * 32 + hiB;
            bfr[nf] = ld_frag(p, p + 16);
        }
        #pragma unroll
        for (int mf = 0; mf < 4; ++mf)
            #pragma unroll
            for (int nf = 0; nf < 2; ++nf)
                acc[mf][nf] = __builtin_amdgcn_wmma_f32_16x16x32_bf16(
                    false, af[mf], false, bfr[nf], (short)0, acc[mf][nf], false, false);
        __syncthreads();
    }

    // ---- epilogue ----
    #pragma unroll
    for (int mf = 0; mf < 4; ++mf) {
        #pragma unroll
        for (int nf = 0; nf < 2; ++nf) {
            const int n = nBase + nW + nf * 16 + l15;
            const float bn = bias[n];
            #pragma unroll
            for (int r = 0; r < 8; ++r) {
                const int m = mBase + mW + mf * 16 + r + rAdd;
                float v = acc[mf][nf][r] + bn;
                if (EPI == 0) {
                    Cf[(size_t)m * N + n] = v;
                } else if (EPI == 1) {
                    Cb[(size_t)m * N + n] = f2bf(v * scale);
                } else if (EPI == 2) {
                    float gel = 0.5f * v * (1.0f + erff(v * 0.70710678118654752f));
                    Cb[(size_t)m * N + n] = f2bf(gel);
                } else {
                    const int bb = m >> 11, l = m & (SEQ - 1);
                    const int hh = n >> 6,  dk = n & (D_KH - 1);
                    Cb[((size_t)((bb * N_HEADS + hh) * D_KH + dk)) * SEQ + l] = f2bf(v);
                }
            }
        }
    }
}

// ---------------------------------------------------------------------------
// Flash attention, bf16 WMMA. One block = (b,h) x 64 query rows.
// 4 waves x 16 q-rows. K tile [kv][dk] and head-transposed V tile [dk][kv]
// streamed by TDM, double-buffered. Q pre-scaled by 1/sqrt(d_k).
// Output in the reference's no-transpose reshape layout.
// ---------------------------------------------------------------------------
__global__ __launch_bounds__(128)
void k_flash_attn(const u16t* __restrict__ Qb, const u16t* __restrict__ Kb,
                  const u16t* __restrict__ Vt, u16t* __restrict__ Ctx)
{
    __shared__ u16t sK[2][64 * 64];     // [kv_row][dk]
    __shared__ u16t sV[2][64 * 64];     // [dk][kv_row]
    __shared__ u16t sP[4][16 * 64];     // per-wave P staging [qrow][kv]

    const int t    = threadIdx.x;
    const int lane = t & 31;
    const int w    = t >> 5;
    const int l15  = lane & 15;
    const int hiA  = (lane < 16) ? 0 : 8;
    const int hiB  = (lane < 16) ? 0 : 8;
    const int rAdd = (lane < 16) ? 0 : 8;

    const int bh = blockIdx.y;
    const int b  = bh >> 4;
    const int h  = bh & 15;
    const int q0 = blockIdx.x * 64 + w * 16;

    const u16t* Kg = Kb + (size_t)(b * SEQ) * D_MODEL + h * D_KH;  // [l][dk] slice
    const u16t* Vg = Vt + (size_t)(bh * D_KH) * SEQ;               // [dk][l] slice

    // Q fragments: 16 rows x 64 (two 16x32 A-frags), loaded once from global.
    v16bf qa[2];
    {
        const size_t rowQ = (size_t)(b * SEQ + q0 + l15) * D_MODEL + h * D_KH;
        #pragma unroll
        for (int kf = 0; kf < 2; ++kf) {
            const u16t* p = Qb + rowQ + kf * 32;
            qa[kf] = ld_frag(p + hiA, p + 16 + hiA);
        }
    }

    v8f o[4];
    #pragma unroll
    for (int i = 0; i < 4; ++i) o[i] = (v8f){0.f,0.f,0.f,0.f,0.f,0.f,0.f,0.f};
    float mrun[8], lrun[8];
    #pragma unroll
    for (int r = 0; r < 8; ++r) { mrun[r] = -3.0e38f; lrun[r] = 0.f; }

    const int NT = SEQ / 64;
    if (w == 0) {
        tdm_load_2d((uint32_t)(uintptr_t)&sK[0][0], Kg, D_MODEL, SEQ, 64, 64, D_MODEL);
        tdm_load_2d((uint32_t)(uintptr_t)&sV[0][0], Vg, SEQ, D_KH, 64, 64, SEQ);
    }

    for (int kt = 0; kt < NT; ++kt) {
        const int cur = kt & 1;
        if (w == 0) {
            if (kt + 1 < NT) {
                const int kv0n = (kt + 1) * 64;
                tdm_load_2d((uint32_t)(uintptr_t)&sK[cur ^ 1][0],
                            Kg + (size_t)kv0n * D_MODEL, D_MODEL, SEQ, 64, 64, D_MODEL);
                tdm_load_2d((uint32_t)(uintptr_t)&sV[cur ^ 1][0],
                            Vg + kv0n, SEQ, D_KH, 64, 64, SEQ);
                __builtin_amdgcn_s_wait_tensorcnt(2);
            } else {
                __builtin_amdgcn_s_wait_tensorcnt(0);
            }
        }
        __syncthreads();

        // ---- S = Q @ K^T : 4 tiles of 16x16 ----
        v8f s[4];
        #pragma unroll
        for (int i = 0; i < 4; ++i) s[i] = (v8f){0.f,0.f,0.f,0.f,0.f,0.f,0.f,0.f};
        #pragma unroll
        for (int nf = 0; nf < 4; ++nf)
            #pragma unroll
            for (int kf = 0; kf < 2; ++kf) {
                const u16t* p = sK[cur] + (nf * 16 + l15) * 64 + kf * 32 + hiB;
                v16bf bk = ld_frag(p, p + 16);
                s[nf] = __builtin_amdgcn_wmma_f32_16x16x32_bf16(
                    false, qa[kf], false, bk, (short)0, s[nf], false, false);
            }

        // ---- online softmax (rows striped across 16-lane halves) ----
        float alpha[8];
        #pragma unroll
        for (int r = 0; r < 8; ++r) {
            float tm = fmaxf(fmaxf(s[0][r], s[1][r]), fmaxf(s[2][r], s[3][r]));
            #pragma unroll
            for (int msk = 1; msk < 16; msk <<= 1)
                tm = fmaxf(tm, __shfl_xor(tm, msk, 32));
            const float mnew = fmaxf(mrun[r], tm);
            alpha[r] = __expf(mrun[r] - mnew);
            mrun[r]  = mnew;
            float rs = 0.f;
            #pragma unroll
            for (int nf = 0; nf < 4; ++nf) {
                float pe = __expf(s[nf][r] - mnew);
                s[nf][r] = pe;
                rs += pe;
            }
            #pragma unroll
            for (int msk = 1; msk < 16; msk <<= 1)
                rs += __shfl_xor(rs, msk, 32);
            lrun[r] = lrun[r] * alpha[r] + rs;
            #pragma unroll
            for (int nf = 0; nf < 4; ++nf) o[nf][r] *= alpha[r];
        }

        // ---- C-layout P -> A-layout via per-wave LDS staging ----
        #pragma unroll
        for (int nf = 0; nf < 4; ++nf)
            #pragma unroll
            for (int r = 0; r < 8; ++r)
                sP[w][(r + rAdd) * 64 + nf * 16 + l15] = f2bf(s[nf][r]);
        __syncthreads();

        // ---- O += P @ V ----
        #pragma unroll
        for (int kf = 0; kf < 2; ++kf) {
            const u16t* pp = sP[w] + l15 * 64 + kf * 32;
            v16bf pa = ld_frag(pp + hiA, pp + 16 + hiA);
            #pragma unroll
            for (int nf = 0; nf < 4; ++nf) {
                const u16t* pv = sV[cur] + (nf * 16 + l15) * 64 + kf * 32 + hiB;
                v16bf bv = ld_frag(pv, pv + 16);
                o[nf] = __builtin_amdgcn_wmma_f32_16x16x32_bf16(
                    false, pa, false, bv, (short)0, o[nf], false, false);
            }
        }
        __syncthreads();   // done with buffers before TDM overwrites next iter
    }

    // ---- normalize and store (faithful no-transpose reshape) ----
    #pragma unroll
    for (int nf = 0; nf < 4; ++nf)
        #pragma unroll
        for (int r = 0; r < 8; ++r) {
            const int ml = q0 + r + rAdd;
            const float v = o[nf][r] / lrun[r];
            Ctx[((size_t)(b * N_HEADS + h) * SEQ + ml) * D_KH + nf * 16 + l15] = f2bf(v);
        }
}

// ---------------------------------------------------------------------------
// Fused residual-add + LayerNorm over d=1024. One block per row, 256 threads.
// ---------------------------------------------------------------------------
template <bool EMIT_BF>
__global__ __launch_bounds__(256)
void k_add_ln(const float* __restrict__ A, const float* __restrict__ R,
              const float* __restrict__ g, const float* __restrict__ beta,
              float* __restrict__ Of, u16t* __restrict__ Ob)
{
    __shared__ float red[8];
    const int row  = blockIdx.x;
    const int t    = threadIdx.x;
    const int lane = t & 31;
    const int w    = t >> 5;
    const size_t base = (size_t)row * D_MODEL + t * 4;

    float4 a = *(const float4*)(A + base);
    float4 r = *(const float4*)(R + base);
    float x0 = a.x + r.x, x1 = a.y + r.y, x2 = a.z + r.z, x3 = a.w + r.w;

    float s = x0 + x1 + x2 + x3;
    #pragma unroll
    for (int m = 1; m < 32; m <<= 1) s += __shfl_xor(s, m, 32);
    if (lane == 0) red[w] = s;
    __syncthreads();
    float mu = 0.f;
    #pragma unroll
    for (int i = 0; i < 8; ++i) mu += red[i];
    mu *= (1.0f / D_MODEL);
    __syncthreads();

    float d0 = x0 - mu, d1 = x1 - mu, d2 = x2 - mu, d3 = x3 - mu;
    float vs = d0 * d0 + d1 * d1 + d2 * d2 + d3 * d3;
    #pragma unroll
    for (int m = 1; m < 32; m <<= 1) vs += __shfl_xor(vs, m, 32);
    if (lane == 0) red[w] = vs;
    __syncthreads();
    float var = 0.f;
    #pragma unroll
    for (int i = 0; i < 8; ++i) var += red[i];
    var *= (1.0f / D_MODEL);
    const float rstd = rsqrtf(var + 1e-5f);

    float4 gv = *(const float4*)(g + t * 4);
    float4 bv = *(const float4*)(beta + t * 4);
    float y0 = d0 * rstd * gv.x + bv.x;
    float y1 = d1 * rstd * gv.y + bv.y;
    float y2 = d2 * rstd * gv.z + bv.z;
    float y3 = d3 * rstd * gv.w + bv.w;
    float4 out = {y0, y1, y2, y3};
    *(float4*)(Of + base) = out;
    if (EMIT_BF) {
        uint2 pk;
        pk.x = (uint32_t)f2bf(y0) | ((uint32_t)f2bf(y1) << 16);
        pk.y = (uint32_t)f2bf(y2) | ((uint32_t)f2bf(y3) << 16);
        *(uint2*)(Ob + base) = pk;
    }
}

// ---------------------------------------------------------------------------
extern "C" void kernel_launch(void* const* d_in, const int* in_sizes, int n_in,
                              void* d_out, int out_size, void* d_ws, size_t ws_size,
                              hipStream_t stream)
{
    const float* X   = (const float*)d_in[0];
    const float* Wq  = (const float*)d_in[1];
    const float* bq  = (const float*)d_in[2];
    const float* Wk  = (const float*)d_in[3];
    const float* bk  = (const float*)d_in[4];
    const float* Wv  = (const float*)d_in[5];
    const float* bv  = (const float*)d_in[6];
    const float* Wo  = (const float*)d_in[7];
    const float* bo  = (const float*)d_in[8];
    const float* W1  = (const float*)d_in[9];
    const float* b1  = (const float*)d_in[10];
    const float* W2  = (const float*)d_in[11];
    const float* b2  = (const float*)d_in[12];
    const float* g1  = (const float*)d_in[13];
    const float* be1 = (const float*)d_in[14];
    const float* g2  = (const float*)d_in[15];
    const float* be2 = (const float*)d_in[16];
    (void)in_sizes; (void)n_in; (void)out_size; (void)ws_size;

    char* p = (char*)d_ws;
    auto take = [&](size_t bytes) -> char* {
        char* r = p;
        p += (bytes + 255) & ~(size_t)255;
        return r;
    };
    u16t* Xb   = (u16t*)take((size_t)SROWS * D_MODEL * 2);
    u16t* Wqt  = (u16t*)take((size_t)D_MODEL * D_MODEL * 2);
    u16t* Wkt  = (u16t*)take((size_t)D_MODEL * D_MODEL * 2);
    u16t* Wvt  = (u16t*)take((size_t)D_MODEL * D_MODEL * 2);
    u16t* Wot  = (u16t*)take((size_t)D_MODEL * D_MODEL * 2);
    u16t* W1t  = (u16t*)take((size_t)D_MODEL * D_FF * 2);
    u16t* W2t  = (u16t*)take((size_t)D_FF * D_MODEL * 2);
    u16t* Qb   = (u16t*)take((size_t)SROWS * D_MODEL * 2);
    u16t* Kbb  = (u16t*)take((size_t)SROWS * D_MODEL * 2);
    u16t* Vt   = (u16t*)take((size_t)SROWS * D_MODEL * 2);
    u16t* Ctx  = (u16t*)take((size_t)SROWS * D_MODEL * 2);
    float* AttnP = (float*)take((size_t)SROWS * D_MODEL * 4);
    float* X1    = (float*)take((size_t)SROWS * D_MODEL * 4);
    u16t*  X1b   = (u16t*)take((size_t)SROWS * D_MODEL * 2);
    u16t*  Hb    = (u16t*)take((size_t)SROWS * D_FF * 2);
    float* FF    = (float*)take((size_t)SROWS * D_MODEL * 4);

    const dim3 blk(256);
    // X stays row-major (A operand); weights converted + transposed (B operand).
    {
        int n4 = (SROWS * D_MODEL) / 4;
        k_cvt4<<<dim3((n4 + 255) / 256), blk, 0, stream>>>(X, Xb, n4);
    }
    k_cvt_t<<<dim3(D_MODEL/32, D_MODEL/32), blk, 0, stream>>>(Wq, Wqt, D_MODEL, D_MODEL);
    k_cvt_t<<<dim3(D_MODEL/32, D_MODEL/32), blk, 0, stream>>>(Wk, Wkt, D_MODEL, D_MODEL);
    k_cvt_t<<<dim3(D_MODEL/32, D_MODEL/32), blk, 0, stream>>>(Wv, Wvt, D_MODEL, D_MODEL);
    k_cvt_t<<<dim3(D_MODEL/32, D_MODEL/32), blk, 0, stream>>>(Wo, Wot, D_MODEL, D_MODEL);
    k_cvt_t<<<dim3(D_FF/32,    D_MODEL/32), blk, 0, stream>>>(W1, W1t, D_MODEL, D_FF);
    k_cvt_t<<<dim3(D_MODEL/32, D_FF/32),    blk, 0, stream>>>(W2, W2t, D_FF, D_MODEL);

    const dim3 gProj(D_MODEL / 128, SROWS / 128);   // (8, 32)
    // QKV projections (Q pre-scaled by 1/sqrt(d_k) = 0.125; V head-transposed)
    k_gemm_bf16<1><<<gProj, blk, 0, stream>>>(Xb, Wqt, bq, nullptr, Qb,
                                              SROWS, D_MODEL, D_MODEL, 0.125f);
    k_gemm_bf16<1><<<gProj, blk, 0, stream>>>(Xb, Wkt, bk, nullptr, Kbb,
                                              SROWS, D_MODEL, D_MODEL, 1.0f);
    k_gemm_bf16<3><<<gProj, blk, 0, stream>>>(Xb, Wvt, bv, nullptr, Vt,
                                              SROWS, D_MODEL, D_MODEL, 1.0f);

    // Flash attention: grid = (q-tiles, b*h)
    const dim3 gAttn(SEQ / 64, BATCH * N_HEADS);    // (32, 32)
    k_flash_attn<<<gAttn, dim3(128), 0, stream>>>(Qb, Kbb, Vt, Ctx);

    // Output projection (f32)
    k_gemm_bf16<0><<<gProj, blk, 0, stream>>>(Ctx, Wot, bo, AttnP, nullptr,
                                              SROWS, D_MODEL, D_MODEL, 1.0f);
    // LN1(attn + x) -> X1 (f32) and X1b (bf16)
    k_add_ln<true><<<dim3(SROWS), blk, 0, stream>>>(AttnP, X, g1, be1, X1, X1b);

    // FFN up + exact GELU -> bf16
    const dim3 gUp(D_FF / 128, SROWS / 128);        // (32, 32)
    k_gemm_bf16<2><<<gUp, blk, 0, stream>>>(X1b, W1t, b1, nullptr, Hb,
                                            SROWS, D_FF, D_MODEL, 1.0f);
    // FFN down -> f32
    k_gemm_bf16<0><<<gProj, blk, 0, stream>>>(Hb, W2t, b2, FF, nullptr,
                                              SROWS, D_MODEL, D_FF, 1.0f);
    // LN2(ff + x1) -> d_out
    k_add_ln<false><<<dim3(SROWS), blk, 0, stream>>>(FF, X1, g2, be2,
                                                     (float*)d_out, nullptr);
}
